// BCRefinement_15934328668762
// MI455X (gfx1250) — matrix operand order repaired and checked by reading, not compiled, not run on hardware
//
#include <hip/hip_runtime.h>
#include <hip/hip_bf16.h>
#include <math.h>

typedef __attribute__((ext_vector_type(16))) _Float16 v16h;
typedef __attribute__((ext_vector_type(2)))  _Float16 h2;
typedef __attribute__((ext_vector_type(8)))  float    v8f;

#define EPSV 1e-6f
#define WPB 4   // waves (tiles) per block for WMMA kernels
#define FRAG_H 512  // halves per fragment record (32 lanes x 16 halves)

static __device__ inline v8f v8zero() {
  v8f z;
#pragma unroll
  for (int i = 0; i < 8; ++i) z[i] = 0.0f;
  return z;
}

static __device__ inline v8f wmma32(v16h a, v16h b, v8f c) {
  // D = A(16x32 f16) * B(32x16 f16) + C(16x16 f32)
  return __builtin_amdgcn_wmma_f32_16x16x32_f16(false, a, false, b, (short)0, c,
                                                false, false);
}

// Inverse of the ISA A-fragment layout (16x32 f16): element (m, kk) -> slot in
// [lane][16 halves] record. lanes 0-15 hold K {0..7,16..23}, lanes 16-31 hold
// K {8..15,24..31}; VGPR j holds a K pair.
static __device__ inline int a_pos(int m, int kk) {
  int lane = m + 16 * ((kk >> 3) & 1);
  int elem = 2 * ((kk >> 4) * 4 + ((kk & 7) >> 1)) + (kk & 1);
  return lane * 16 + elem;
}

// Inverse of the ISA B-fragment layout (32x16 f16): element (kk, n) -> slot.
// lanes 0-15: column n, K {0..7,16..23}; lanes 16-31: column n, K {8..15,24..31}.
static __device__ inline int b_pos(int kk, int n) {
  return (n + 16 * ((kk >> 3) & 1)) * 16 + 8 * (kk >> 4) + (kk & 7);
}

// Load one fragment: 32 contiguous bytes per lane -> 2x ds_load_b128.
static __device__ inline v16h frag_ld(const _Float16* s) {
  return *(const v16h*)(s + (threadIdx.x & 31) * 16);
}

// Stage weights W[K][C] (row-major f32 in global) into fragment-swizzled f16 LDS.
// Fragment f = (kc * (C/16) + nt) occupies FRAG_H halves.
static __device__ inline void stage_w(_Float16* dst, const float* __restrict__ src,
                                      int K, int C) {
  const int NT = C >> 4;
  for (int i = threadIdx.x; i < K * C; i += blockDim.x) {
    int k = i / C, c = i - k * C;
    dst[((k >> 5) * NT + (c >> 4)) * FRAG_H + b_pos(k & 31, c & 15)] =
        (_Float16)src[i];
  }
}

static __device__ inline float sigm(float x) { return 1.0f / (1.0f + __expf(-x)); }

// ---------------------------------------------------------------- utility kernels
__global__ void copy_f32_kernel(float* __restrict__ d, const float* __restrict__ s,
                                long long n) {
  for (long long i = (long long)blockIdx.x * blockDim.x + threadIdx.x; i < n;
       i += (long long)gridDim.x * blockDim.x)
    d[i] = s[i];
}

__global__ void fill_f32_kernel(float* __restrict__ p, float v, long long n) {
  for (long long i = (long long)blockIdx.x * blockDim.x + threadIdx.x; i < n;
       i += (long long)gridDim.x * blockDim.x)
    p[i] = v;
}

__global__ void fill_i32_kernel(int* __restrict__ p, int v, long long n) {
  for (long long i = (long long)blockIdx.x * blockDim.x + threadIdx.x; i < n;
       i += (long long)gridDim.x * blockDim.x)
    p[i] = v;
}

// ---------------------------------------------------------------- precompute
__global__ void mark_cpl_kernel(const int* __restrict__ row, const int* __restrict__ col,
                                const int* __restrict__ tie, int* __restrict__ iscpl,
                                float* __restrict__ degv, int E) {
  int e = blockIdx.x * blockDim.x + threadIdx.x;
  if (e >= E) return;
  if (tie[e]) {
    atomicOr(&iscpl[row[e]], 1);
    atomicOr(&iscpl[col[e]], 1);
    atomicAdd(&degv[row[e]], 1.0f);
    atomicAdd(&degv[col[e]], 1.0f);
  }
}

__global__ void edge_w_kernel(const int* __restrict__ col, const int* __restrict__ tie,
                              const int* __restrict__ iscpl, const float* __restrict__ ear,
                              float* __restrict__ wraw, float* __restrict__ wsum, int E) {
  int e = blockIdx.x * blockDim.x + threadIdx.x;
  if (e >= E) return;
  float X  = fabsf(ear[(size_t)e * 10 + 1]);
  float to = (!tie[e] && iscpl[col[e]]) ? 1.0f : 0.0f;
  float w  = to * (1.0f / sqrtf(X * X + EPSV));
  wraw[e]  = w;
  if (w != 0.0f) atomicAdd(&wsum[col[e]], w);
}

__global__ void edge_wn_kernel(const int* __restrict__ col, float* __restrict__ wraw,
                               const float* __restrict__ wsum, int E) {
  int e = blockIdx.x * blockDim.x + threadIdx.x;
  if (e >= E) return;
  wraw[e] = wraw[e] / (wsum[col[e]] + EPSV);
}

// ---------------------------------------------------------------- edge MLP (WMMA)
// e_work[e] += tie[e] * MLP2([e_work | h[row] | h[col]])   (16 edges per wave)
__global__ void __launch_bounds__(32 * WPB, 1)
edge_mlp_kernel(float* __restrict__ eW, const float* __restrict__ hW,
                const int* __restrict__ row, const int* __restrict__ col,
                const int* __restrict__ tie, const float* __restrict__ We1,
                const float* __restrict__ be1, const float* __restrict__ We2,
                const float* __restrict__ be2, int E, int Et) {
  extern __shared__ char smem[];
  _Float16* sW1 = (_Float16*)smem;            // 48 frags (K=192, C=128)
  _Float16* sW2 = sW1 + 48 * FRAG_H;          // 16 frags (K=128, C=64)
  float*    sb1 = (float*)(sW2 + 16 * FRAG_H);
  float*    sb2 = sb1 + 128;
  char*   wbase = (char*)(sb2 + 64);
  const int wave = threadIdx.x >> 5, lane = threadIdx.x & 31;
  const size_t PER = (size_t)(6 + 4) * FRAG_H * 2 + 64;  // sA(6f) + sH(4f) + tie
  _Float16* sA   = (_Float16*)(wbase + (size_t)wave * PER);
  _Float16* sH   = sA + 6 * FRAG_H;
  float*    sTie = (float*)(sH + 4 * FRAG_H);

  stage_w(sW1, We1, 192, 128);
  stage_w(sW2, We2, 128, 64);
  for (int i = threadIdx.x; i < 128; i += blockDim.x) sb1[i] = be1[i];
  for (int i = threadIdx.x; i < 64; i += blockDim.x)  sb2[i] = be2[i];
  __syncthreads();

  const int n = lane & 15, hi = lane >> 4;
  for (int tile = blockIdx.x * WPB + wave; tile < Et; tile += gridDim.x * WPB) {
    const int base = tile * 16;
    {
      int nt = tile + gridDim.x * WPB;
      if (nt < Et) __builtin_prefetch(&eW[(size_t)nt * 16 * 64 + lane * 32], 0, 1);
    }
    // stage A = [e | h_src | h_dst] (f16, fragment-swizzled, paired stores)
    for (int i = lane; i < 16 * 96; i += 32) {
      int m = i / 96, cp = (i - m * 96) * 2;
      int e = base + m;
      float v0 = 0.0f, v1 = 0.0f;
      if (e < E) {
        const float* s;
        if (cp < 64)       s = &eW[(size_t)e * 64 + cp];
        else if (cp < 128) s = &hW[(size_t)row[e] * 64 + (cp - 64)];
        else               s = &hW[(size_t)col[e] * 64 + (cp - 128)];
        v0 = s[0]; v1 = s[1];
      }
      h2 pr; pr[0] = (_Float16)v0; pr[1] = (_Float16)v1;
      *(h2*)(sA + (cp >> 5) * FRAG_H + a_pos(m, cp & 31)) = pr;
    }
    if (lane < 16) sTie[lane] = (base + lane < E) ? (float)tie[base + lane] : 0.0f;
    __builtin_amdgcn_wave_barrier();

    // layer 1: (16x192) @ (192x128) -> 8 n-tiles, 6 k-chunks
    v8f acc[8];
#pragma unroll
    for (int t = 0; t < 8; ++t) acc[t] = v8zero();
#pragma unroll
    for (int kc = 0; kc < 6; ++kc) {
      v16h a = frag_ld(sA + kc * FRAG_H);
#pragma unroll
      for (int t = 0; t < 8; ++t)
        acc[t] = wmma32(a, frag_ld(sW1 + (kc * 8 + t) * FRAG_H), acc[t]);
    }
    // bias + ReLU -> LDS (fragment-swizzled hidden)
#pragma unroll
    for (int t = 0; t < 8; ++t) {
      float bv = sb1[16 * t + n];
#pragma unroll
      for (int v = 0; v < 8; ++v) {
        float x = acc[t][v] + bv;
        int c = 16 * t + n;
        sH[(c >> 5) * FRAG_H + a_pos(v + 8 * hi, c & 31)] =
            (_Float16)(x > 0.0f ? x : 0.0f);
      }
    }
    __builtin_amdgcn_wave_barrier();

    // layer 2: (16x128) @ (128x64) -> 4 n-tiles, 4 k-chunks
    v8f acc2[4];
#pragma unroll
    for (int t = 0; t < 4; ++t) acc2[t] = v8zero();
#pragma unroll
    for (int kc = 0; kc < 4; ++kc) {
      v16h a = frag_ld(sH + kc * FRAG_H);
#pragma unroll
      for (int t = 0; t < 4; ++t)
        acc2[t] = wmma32(a, frag_ld(sW2 + (kc * 4 + t) * FRAG_H), acc2[t]);
    }
    // masked residual in fp32
#pragma unroll
    for (int t = 0; t < 4; ++t) {
      float bv = sb2[16 * t + n];
#pragma unroll
      for (int v = 0; v < 8; ++v) {
        int m = v + 8 * hi, e = base + m;
        if (e < E) {
          size_t off = (size_t)e * 64 + 16 * t + n;
          eW[off] = eW[off] + sTie[m] * (acc2[t][v] + bv);
        }
      }
    }
  }
}

// ---------------------------------------------------------------- hm = h @ W_nn + b_nn
__global__ void __launch_bounds__(32 * WPB, 1)
node_gemm_kernel(const float* __restrict__ hW, const float* __restrict__ Wnn,
                 const float* __restrict__ bnn, float* __restrict__ hm, int N, int Nt) {
  extern __shared__ char smem[];
  _Float16* sW = (_Float16*)smem;          // 8 frags (K=64, C=64)
  float*    sb = (float*)(sW + 8 * FRAG_H);
  char*  wbase = (char*)(sb + 64);
  const int wave = threadIdx.x >> 5, lane = threadIdx.x & 31;
  _Float16* sA = (_Float16*)(wbase + (size_t)wave * (2 * FRAG_H * 2));

  stage_w(sW, Wnn, 64, 64);
  for (int i = threadIdx.x; i < 64; i += blockDim.x) sb[i] = bnn[i];
  __syncthreads();

  const int n = lane & 15, hi = lane >> 4;
  for (int tile = blockIdx.x * WPB + wave; tile < Nt; tile += gridDim.x * WPB) {
    const int base = tile * 16;
    for (int i = lane; i < 16 * 32; i += 32) {
      int m = i >> 5, cp = (i & 31) * 2, idx = base + m;
      float v0 = 0.0f, v1 = 0.0f;
      if (idx < N) {
        const float* s = &hW[(size_t)idx * 64 + cp];
        v0 = s[0]; v1 = s[1];
      }
      h2 pr; pr[0] = (_Float16)v0; pr[1] = (_Float16)v1;
      *(h2*)(sA + (cp >> 5) * FRAG_H + a_pos(m, cp & 31)) = pr;
    }
    __builtin_amdgcn_wave_barrier();
    v8f acc[4];
#pragma unroll
    for (int t = 0; t < 4; ++t) acc[t] = v8zero();
#pragma unroll
    for (int kc = 0; kc < 2; ++kc) {
      v16h a = frag_ld(sA + kc * FRAG_H);
#pragma unroll
      for (int t = 0; t < 4; ++t)
        acc[t] = wmma32(a, frag_ld(sW + (kc * 4 + t) * FRAG_H), acc[t]);
    }
#pragma unroll
    for (int t = 0; t < 4; ++t) {
      float bv = sb[16 * t + n];
#pragma unroll
      for (int v = 0; v < 8; ++v) {
        int m = v + 8 * hi, idx = base + m;
        if (idx < N) hm[(size_t)idx * 64 + 16 * t + n] = acc[t][v] + bv;
      }
    }
    __builtin_amdgcn_wave_barrier();
  }
}

// ---------------------------------------------------------------- scatter (atomics)
__global__ void scatter_kernel(const float* __restrict__ hm, const float* __restrict__ eW,
                               const int* __restrict__ row, const int* __restrict__ col,
                               const int* __restrict__ tie, const float* __restrict__ wn,
                               float* __restrict__ mnode, float* __restrict__ medge,
                               long long total) {
  long long id = (long long)blockIdx.x * blockDim.x + threadIdx.x;
  if (id >= total) return;
  int e = (int)(id >> 6), c = (int)(id & 63);
  float w = wn[e];
  if (w != 0.0f)
    atomicAdd(&mnode[(size_t)col[e] * 64 + c], hm[(size_t)row[e] * 64 + c] * w);
  if (tie[e]) {
    float ev = eW[(size_t)e * 64 + c];
    atomicAdd(&medge[(size_t)row[e] * 64 + c], ev);
    atomicAdd(&medge[(size_t)col[e] * 64 + c], ev);
  }
}

// ---------------------------------------------------------------- gate + GRU (WMMA)
__global__ void __launch_bounds__(32 * WPB, 1)
node_update_kernel(float* __restrict__ hW, const float* __restrict__ medge,
                   const float* __restrict__ mnode, const float* __restrict__ degv,
                   const int* __restrict__ iscpl, const float* __restrict__ Wg1,
                   const float* __restrict__ bg1, const float* __restrict__ Wg2,
                   const float* __restrict__ bg2, const float* __restrict__ Wih,
                   const float* __restrict__ bih, const float* __restrict__ Whh,
                   const float* __restrict__ bhh, int N, int Nt) {
  extern __shared__ char smem[];
  _Float16* sWg1 = (_Float16*)smem;             // 24 frags (K=192, C=64)
  _Float16* sWih = sWg1 + 24 * FRAG_H;          // 24 frags (K=64, C=192)
  _Float16* sWhh = sWih + 24 * FRAG_H;          // 24 frags
  float* sWg2 = (float*)(sWhh + 24 * FRAG_H);   // 64
  float* sbg1 = sWg2 + 64;                      // 64
  float* sbih = sbg1 + 64;                      // 192
  float* sbhh = sbih + 192;                     // 192
  char* wbase = (char*)(sbhh + 192);
  const int wave = threadIdx.x >> 5, lane = threadIdx.x & 31;
  const size_t PER = (size_t)6 * FRAG_H * 2 + 128;  // sA(6 frags) + gate + cpl
  _Float16* sA  = (_Float16*)(wbase + (size_t)wave * PER); // [me|mn|h] swizzled
  float* sGate  = (float*)(sA + 6 * FRAG_H);               // 16
  float* sCpl   = sGate + 16;                              // 16

  stage_w(sWg1, Wg1, 192, 64);
  stage_w(sWih, Wih, 64, 192);
  stage_w(sWhh, Whh, 64, 192);
  for (int i = threadIdx.x; i < 64; i += blockDim.x) { sWg2[i] = Wg2[i]; sbg1[i] = bg1[i]; }
  for (int i = threadIdx.x; i < 192; i += blockDim.x) { sbih[i] = bih[i]; sbhh[i] = bhh[i]; }
  __syncthreads();

  const int n = lane & 15, hi = lane >> 4;
  const float bg2v = bg2[0];
  for (int tile = blockIdx.x * WPB + wave; tile < Nt; tile += gridDim.x * WPB) {
    const int base = tile * 16;
    for (int i = lane; i < 16 * 96; i += 32) {
      int m = i / 96, cp = (i - m * 96) * 2, idx = base + m;
      float v0 = 0.0f, v1 = 0.0f;
      if (idx < N) {
        if (cp < 64) {
          float dv = fmaxf(degv[idx], 1.0f);
          const float* s = &medge[(size_t)idx * 64 + cp];
          v0 = s[0] / dv; v1 = s[1] / dv;
        } else if (cp < 128) {
          const float* s = &mnode[(size_t)idx * 64 + (cp - 64)];
          v0 = s[0]; v1 = s[1];
        } else {
          const float* s = &hW[(size_t)idx * 64 + (cp - 128)];
          v0 = s[0]; v1 = s[1];
        }
      }
      h2 pr; pr[0] = (_Float16)v0; pr[1] = (_Float16)v1;
      *(h2*)(sA + (cp >> 5) * FRAG_H + a_pos(m, cp & 31)) = pr;
    }
    if (lane < 16) sCpl[lane] = (base + lane < N) ? (float)iscpl[base + lane] : 0.0f;
    __builtin_amdgcn_wave_barrier();

    // gate hidden: (16x192) @ (192x64), ReLU, dot with W_g2 via shuffle reduce
    v8f g[4];
#pragma unroll
    for (int t = 0; t < 4; ++t) g[t] = v8zero();
#pragma unroll
    for (int kc = 0; kc < 6; ++kc) {
      v16h a = frag_ld(sA + kc * FRAG_H);
#pragma unroll
      for (int t = 0; t < 4; ++t)
        g[t] = wmma32(a, frag_ld(sWg1 + (kc * 4 + t) * FRAG_H), g[t]);
    }
    float part[8];
#pragma unroll
    for (int v = 0; v < 8; ++v) part[v] = 0.0f;
#pragma unroll
    for (int t = 0; t < 4; ++t) {
      float bv = sbg1[16 * t + n], wv = sWg2[16 * t + n];
#pragma unroll
      for (int v = 0; v < 8; ++v) {
        float x = g[t][v] + bv;
        part[v] += (x > 0.0f ? x : 0.0f) * wv;
      }
    }
#pragma unroll
    for (int off = 1; off < 16; off <<= 1)
#pragma unroll
      for (int v = 0; v < 8; ++v) part[v] += __shfl_xor(part[v], off, 32);
    if (n == 0)
#pragma unroll
      for (int v = 0; v < 8; ++v) sGate[v + 8 * hi] = sigm(part[v] + bg2v);
    __builtin_amdgcn_wave_barrier();

    // m = m_edge + gate * m_node (paired, swizzled positions align: f vs f+2)
    for (int i = lane; i < FRAG_H; i += 32) {
      int f = i >> 8, p = (i & 255) * 2;
      int m = (p >> 4) & 15;
      h2 me2 = *(h2*)(sA + f * FRAG_H + p);
      h2 mn2 = *(h2*)(sA + (f + 2) * FRAG_H + p);
      float gt = sGate[m];
      h2 r;
      r[0] = (_Float16)((float)me2[0] + gt * (float)mn2[0]);
      r[1] = (_Float16)((float)me2[1] + gt * (float)mn2[1]);
      *(h2*)(sA + f * FRAG_H + p) = r;
    }
    __builtin_amdgcn_wave_barrier();

    v16h am0 = frag_ld(sA + 0 * FRAG_H), am1 = frag_ld(sA + 1 * FRAG_H);
    v16h ah0 = frag_ld(sA + 4 * FRAG_H), ah1 = frag_ld(sA + 5 * FRAG_H);

    for (int t = 0; t < 4; ++t) {
      v8f giR = v8zero(), giZ = v8zero(), giN = v8zero();
      v8f ghR = v8zero(), ghZ = v8zero(), ghN = v8zero();
      giR = wmma32(am0, frag_ld(sWih + (0 * 12 + t) * FRAG_H), giR);
      giR = wmma32(am1, frag_ld(sWih + (1 * 12 + t) * FRAG_H), giR);
      giZ = wmma32(am0, frag_ld(sWih + (0 * 12 + 4 + t) * FRAG_H), giZ);
      giZ = wmma32(am1, frag_ld(sWih + (1 * 12 + 4 + t) * FRAG_H), giZ);
      giN = wmma32(am0, frag_ld(sWih + (0 * 12 + 8 + t) * FRAG_H), giN);
      giN = wmma32(am1, frag_ld(sWih + (1 * 12 + 8 + t) * FRAG_H), giN);
      ghR = wmma32(ah0, frag_ld(sWhh + (0 * 12 + t) * FRAG_H), ghR);
      ghR = wmma32(ah1, frag_ld(sWhh + (1 * 12 + t) * FRAG_H), ghR);
      ghZ = wmma32(ah0, frag_ld(sWhh + (0 * 12 + 4 + t) * FRAG_H), ghZ);
      ghZ = wmma32(ah1, frag_ld(sWhh + (1 * 12 + 4 + t) * FRAG_H), ghZ);
      ghN = wmma32(ah0, frag_ld(sWhh + (0 * 12 + 8 + t) * FRAG_H), ghN);
      ghN = wmma32(ah1, frag_ld(sWhh + (1 * 12 + 8 + t) * FRAG_H), ghN);

      float biR = sbih[16 * t + n],       bhR = sbhh[16 * t + n];
      float biZ = sbih[64 + 16 * t + n],  bhZ = sbhh[64 + 16 * t + n];
      float biN = sbih[128 + 16 * t + n], bhN = sbhh[128 + 16 * t + n];
#pragma unroll
      for (int v = 0; v < 8; ++v) {
        int m = v + 8 * hi, idx = base + m;
        float r  = sigm(giR[v] + biR + ghR[v] + bhR);
        float z  = sigm(giZ[v] + biZ + ghZ[v] + bhZ);
        float nn = tanhf(giN[v] + biN + r * (ghN[v] + bhN));
        if (idx < N && sCpl[m] > 0.0f) {
          size_t off = (size_t)idx * 64 + 16 * t + n;
          float hold = hW[off];
          hW[off] = (1.0f - z) * nn + z * hold;
        }
      }
    }
  }
}

// ---------------------------------------------------------------- host launcher
extern "C" void kernel_launch(void* const* d_in, const int* in_sizes, int n_in,
                              void* d_out, int out_size, void* d_ws, size_t ws_size,
                              hipStream_t stream) {
  const float* h_in = (const float*)d_in[0];
  const float* e_in = (const float*)d_in[1];
  const int* eidx   = (const int*)d_in[2];
  const int* tie    = (const int*)d_in[3];
  const float* ear  = (const float*)d_in[4];
  const float* We1  = (const float*)d_in[5];  const float* be1 = (const float*)d_in[6];
  const float* We2  = (const float*)d_in[7];  const float* be2 = (const float*)d_in[8];
  const float* Wih  = (const float*)d_in[9];  const float* bih = (const float*)d_in[10];
  const float* Whh  = (const float*)d_in[11]; const float* bhh = (const float*)d_in[12];
  const float* Wnn  = (const float*)d_in[13]; const float* bnn = (const float*)d_in[14];
  const float* Wg1  = (const float*)d_in[15]; const float* bg1 = (const float*)d_in[16];
  const float* Wg2  = (const float*)d_in[17]; const float* bg2 = (const float*)d_in[18];

  const int N = in_sizes[0] / 64;
  const int E = in_sizes[1] / 64;
  const int* row = eidx;
  const int* col = eidx + E;

  float* hW = (float*)d_out;              // [N][64]
  float* eW = hW + (size_t)N * 64;        // [E][64]

  char* p = (char*)d_ws;
  auto carve = [&](size_t bytes) {
    char* r = p;
    p += (bytes + 255) & ~(size_t)255;
    return r;
  };
  float* wnorm = (float*)carve((size_t)E * 4);
  int*   iscpl = (int*)carve((size_t)N * 4);
  float* wsum  = (float*)carve((size_t)N * 4);
  float* degv  = (float*)carve((size_t)N * 4);
  float* hm    = (float*)carve((size_t)N * 64 * 4);
  float* mnode = (float*)carve((size_t)N * 64 * 4);
  float* medge = (float*)carve((size_t)N * 64 * 4);

  const int Et = (E + 15) / 16, Nt = (N + 15) / 16;
  const long long nh = (long long)N * 64, ne = (long long)E * 64;

  copy_f32_kernel<<<4096, 256, 0, stream>>>(hW, h_in, nh);
  copy_f32_kernel<<<4096, 256, 0, stream>>>(eW, e_in, ne);
  fill_i32_kernel<<<1024, 256, 0, stream>>>(iscpl, 0, N);
  fill_f32_kernel<<<1024, 256, 0, stream>>>(wsum, 0.0f, N);
  fill_f32_kernel<<<1024, 256, 0, stream>>>(degv, 0.0f, N);

  const int eb = (E + 255) / 256;
  mark_cpl_kernel<<<eb, 256, 0, stream>>>(row, col, tie, iscpl, degv, E);
  edge_w_kernel<<<eb, 256, 0, stream>>>(col, tie, iscpl, ear, wnorm, wsum, E);
  edge_wn_kernel<<<eb, 256, 0, stream>>>(col, wnorm, wsum, E);

  const size_t smem_edge = (size_t)(48 + 16) * FRAG_H * 2 + (128 + 64) * 4 +
                           (size_t)WPB * ((size_t)(6 + 4) * FRAG_H * 2 + 64);
  const size_t smem_ng = (size_t)8 * FRAG_H * 2 + 64 * 4 +
                         (size_t)WPB * ((size_t)2 * FRAG_H * 2);
  const size_t smem_nu = (size_t)72 * FRAG_H * 2 + (64 + 64 + 192 + 192) * 4 +
                         (size_t)WPB * ((size_t)6 * FRAG_H * 2 + 128);

  const int egrid = (Et + WPB - 1) / WPB;
  const int ngrid = (Nt + WPB - 1) / WPB;
  const long long sc_total = ne;
  const int sgrid = (int)((sc_total + 255) / 256);

  for (int it = 0; it < 2; ++it) {
    edge_mlp_kernel<<<egrid, 32 * WPB, smem_edge, stream>>>(
        eW, hW, row, col, tie, We1, be1, We2, be2, E, Et);
    fill_f32_kernel<<<4096, 256, 0, stream>>>(mnode, 0.0f, nh);
    fill_f32_kernel<<<4096, 256, 0, stream>>>(medge, 0.0f, nh);
    node_gemm_kernel<<<ngrid, 32 * WPB, smem_ng, stream>>>(hW, Wnn, bnn, hm, N, Nt);
    scatter_kernel<<<sgrid, 256, 0, stream>>>(hm, eW, row, col, tie, wnorm, mnode,
                                              medge, sc_total);
    node_update_kernel<<<ngrid, 32 * WPB, smem_nu, stream>>>(
        hW, medge, mnode, degv, iscpl, Wg1, bg1, Wg2, bg2, Wih, bih, Whh, bhh, N, Nt);
  }
}
